// GNOBlockSingleConv_30494267802196
// MI455X (gfx1250) — matrix operand order repaired and checked by reading, not compiled, not run on hardware
//
#include <hip/hip_runtime.h>
#include <hip/hip_bf16.h>

#define NNODES 4096
#define NEDGES 65536
#define LDIM   64
#define KDIM   256

typedef unsigned short u16;
typedef unsigned int   u32;

typedef __attribute__((ext_vector_type(16))) __bf16 v16bf;
typedef __attribute__((ext_vector_type(8)))  float  v8f;

union Frag16 { v16bf v; uint4 q[2]; u16 h[16]; };

__device__ __forceinline__ float gelu_f(float x) {
    return 0.5f * x * (1.0f + erff(x * 0.7071067811865476f));
}
__device__ __forceinline__ u16 f2bf(float f) {
    u32 u = __float_as_uint(f);
    return (u16)((u + 0x7FFFu + ((u >> 16) & 1u)) >> 16);   // RNE
}

// ---------------------------------------------------------------------------
// Convert + transpose W2 [256,256] and W3 [256,4096] (fp32, [k,n]) into
// bf16 [n,k] layouts so WMMA B-fragments are contiguous-K b128 loads.
// ---------------------------------------------------------------------------
__global__ void convert_weights_kernel(const float* __restrict__ W2,
                                       const float* __restrict__ W3,
                                       u16* __restrict__ W2t,
                                       u16* __restrict__ W3t) {
    const int stride = gridDim.x * blockDim.x;
    for (int i = blockIdx.x * blockDim.x + threadIdx.x; i < 256 * 4096; i += stride) {
        int k = i >> 12, n = i & 4095;
        W3t[n * 256 + k] = f2bf(W3[i]);
    }
    for (int i = blockIdx.x * blockDim.x + threadIdx.x; i < 256 * 256; i += stride) {
        int k = i >> 8, n = i & 255;
        W2t[n * 256 + k] = f2bf(W2[i]);
    }
}

// Load B fragment for n-tile J of this k-step (fixed immediate offsets).
#define LOAD_B(J, BV)                                                           \
    {                                                                           \
        const u16* bp = Bptr + k0 + (J) * 4096;                                 \
        BV.q[0] = *(const uint4*)bp;                                            \
        BV.q[1] = *(const uint4*)(bp + 8);                                      \
    }

#define MAC(BV, ACC)                                                            \
    ACC = __builtin_amdgcn_wmma_f32_16x16x32_bf16(                              \
              false, a.v, false, BV.v, (short)0, ACC, false, false);

// Inner k-step: A frag from LDS, all 8 B frags issued up front (one clause),
// then the 8-WMMA chain with staggered load waits.
#define K_STEP                                                                  \
    {                                                                           \
        Frag16 a;                                                               \
        const u16* ap = &tile[lm * 264 + k0 + kselA];                           \
        a.q[0] = *(const uint4*)ap;                                             \
        a.q[1] = *(const uint4*)(ap + 8);                                       \
        Frag16 f0, f1, f2, f3, f4, f5, f6, f7;                                  \
        LOAD_B(0, f0) LOAD_B(1, f1) LOAD_B(2, f2) LOAD_B(3, f3)                 \
        LOAD_B(4, f4) LOAD_B(5, f5) LOAD_B(6, f6) LOAD_B(7, f7)                 \
        MAC(f0, acc0) MAC(f1, acc1) MAC(f2, acc2) MAC(f3, acc3)                 \
        MAC(f4, acc4) MAC(f5, acc5) MAC(f6, acc6) MAC(f7, acc7)                 \
    }

// Write one 16x16 accumulator tile (lane's 8 values) into LDS D tile.
#define STORE_ACC_GELU(J, ACC)                                                  \
    {                                                                           \
        const int col = (nbase + (J)) * 16 + ncol;                              \
        const float bv = b2[col];                                               \
        _Pragma("unroll")                                                       \
        for (int r = 0; r < 8; r++)                                             \
            Ds[(mbase + r) * 264 + col] = f2bf(gelu_f(ACC[r] + bv));            \
    }

#define STORE_ACC_BIAS(J, ACC)                                                  \
    {                                                                           \
        const int col = (nbase + (J)) * 16 + ncol;                              \
        const float bv = b3[nblk + col];                                        \
        _Pragma("unroll")                                                       \
        for (int r = 0; r < 8; r++)                                             \
            Ds[(mbase + r) * 264 + col] = f2bf(ACC[r] + bv);                    \
    }

// ---------------------------------------------------------------------------
// Edge MLP layers 1+2 fused:  h2 = GELU(GELU(ea@W1+b1)@W2+b2)  -> bf16 [E,256]
// Block = 64 edges, 256 threads (8 waves). Layer1 scalar -> LDS bf16 tile,
// layer2 = bf16 WMMA, each wave computes a 16x128 strip.
// ---------------------------------------------------------------------------
__global__ void __launch_bounds__(256) mlp12_kernel(
    const float* __restrict__ edge_attr,
    const float* __restrict__ W1, const float* __restrict__ b1,
    const u16*  __restrict__ W2t, const float* __restrict__ b2,
    u16* __restrict__ h2)
{
    __shared__ __align__(16) float eas[64 * 6];
    __shared__ __align__(16) u16   tile[64 * 264];   // A tile, reused for D tile

    const int tid = threadIdx.x;
    const int e0  = blockIdx.x * 64;

    if (tid < 64 * 6) eas[tid] = edge_attr[e0 * 6 + tid];
    __syncthreads();

    // ---- layer 1: thread `tid` produces column `tid` of h1 for all 64 rows
    {
        const float w0 = W1[0 * 256 + tid], w1 = W1[1 * 256 + tid], w2 = W1[2 * 256 + tid];
        const float w3 = W1[3 * 256 + tid], w4 = W1[4 * 256 + tid], w5 = W1[5 * 256 + tid];
        const float bb = b1[tid];
        #pragma unroll 4
        for (int r = 0; r < 64; r++) {
            const float* ea = &eas[r * 6];
            float s = bb + ea[0]*w0 + ea[1]*w1 + ea[2]*w2 + ea[3]*w3 + ea[4]*w4 + ea[5]*w5;
            tile[r * 264 + tid] = f2bf(gelu_f(s));
        }
    }
    __syncthreads();

    // ---- layer 2: h2[64,256] = gelu(h1 @ W2 + b2) via WMMA bf16
    const int lane  = tid & 31;
    const int wav   = tid >> 5;
    const int mtile = wav >> 1;          // 0..3
    const int nbase = (wav & 1) * 8;     // 8 n-tiles per wave

    v8f acc0 = {}, acc1 = {}, acc2 = {}, acc3 = {};
    v8f acc4 = {}, acc5 = {}, acc6 = {}, acc7 = {};

    const int lm    = mtile * 16 + (lane & 15);
    const int kselA = (lane >> 4) ? 16 : 0;
    const int kselB = (lane >> 4) ? 16 : 0;
    const int ncol  = lane & 15;
    const u16* Bptr = &W2t[(size_t)(nbase * 16 + ncol) * 256 + kselB];

    #pragma unroll 1
    for (int ks = 0; ks < 8; ks++) {
        const int k0 = ks * 32;
        K_STEP
    }

    __syncthreads();                       // done reading A tile
    u16* Ds = tile;                        // reuse as D tile
    const int mbase = mtile * 16 + ((lane >> 4) ? 8 : 0);
    STORE_ACC_GELU(0, acc0)  STORE_ACC_GELU(1, acc1)
    STORE_ACC_GELU(2, acc2)  STORE_ACC_GELU(3, acc3)
    STORE_ACC_GELU(4, acc4)  STORE_ACC_GELU(5, acc5)
    STORE_ACC_GELU(6, acc6)  STORE_ACC_GELU(7, acc7)
    __syncthreads();

    // coalesced b128 store of the 64x256 bf16 D tile
    #pragma unroll
    for (int i = 0; i < 8; i++) {
        const int idx = tid + i * 256;
        const int r = idx >> 5;
        const int c = (idx & 31) * 8;
        *(uint4*)&h2[(size_t)(e0 + r) * 256 + c] = *(const uint4*)&Ds[r * 264 + c];
    }
}

// ---------------------------------------------------------------------------
// W_edge (bf16 [E,4096]) = h2 @ W3 + b3.  M=65536, N=4096, K=256.
// Block tile 64x256, A staged in LDS, B from transposed bf16 W3 (L2-resident).
// ---------------------------------------------------------------------------
__global__ void __launch_bounds__(256) wedge_gemm_kernel(
    const u16* __restrict__ h2, const u16* __restrict__ W3t,
    const float* __restrict__ b3, u16* __restrict__ Wedge)
{
    __shared__ __align__(16) u16 tile[64 * 264];   // A tile, reused for D tile

    const int tid  = threadIdx.x;
    const int e0   = blockIdx.x * 64;
    const int nblk = blockIdx.y * 256;

    // stage A tile [64,256] bf16 (2048 uint4 / 256 threads = 8 each)
    #pragma unroll
    for (int i = 0; i < 8; i++) {
        const int idx = tid + i * 256;
        const int r = idx >> 5;
        const int c = (idx & 31) * 8;
        *(uint4*)&tile[r * 264 + c] = *(const uint4*)&h2[(size_t)(e0 + r) * 256 + c];
    }
    __syncthreads();

    const int lane  = tid & 31;
    const int wav   = tid >> 5;
    const int mtile = wav >> 1;
    const int nbase = (wav & 1) * 8;

    v8f acc0 = {}, acc1 = {}, acc2 = {}, acc3 = {};
    v8f acc4 = {}, acc5 = {}, acc6 = {}, acc7 = {};

    const int lm    = mtile * 16 + (lane & 15);
    const int kselA = (lane >> 4) ? 16 : 0;
    const int kselB = (lane >> 4) ? 16 : 0;
    const int ncol  = lane & 15;
    const u16* Bptr = &W3t[(size_t)(nblk + nbase * 16 + ncol) * 256 + kselB];

    #pragma unroll 1
    for (int ks = 0; ks < 8; ks++) {
        const int k0 = ks * 32;
        K_STEP
    }

    __syncthreads();                       // done reading A tile
    u16* Ds = tile;                        // reuse as D tile
    const int mbase = mtile * 16 + ((lane >> 4) ? 8 : 0);
    STORE_ACC_BIAS(0, acc0)  STORE_ACC_BIAS(1, acc1)
    STORE_ACC_BIAS(2, acc2)  STORE_ACC_BIAS(3, acc3)
    STORE_ACC_BIAS(4, acc4)  STORE_ACC_BIAS(5, acc5)
    STORE_ACC_BIAS(6, acc6)  STORE_ACC_BIAS(7, acc7)
    __syncthreads();

    // coalesced b128 store of the 64x256 bf16 D tile
    #pragma unroll
    for (int i = 0; i < 8; i++) {
        const int idx = tid + i * 256;
        const int r = idx >> 5;
        const int c = (idx & 31) * 8;
        *(uint4*)&Wedge[(size_t)(e0 + r) * 4096 + nblk + c] = *(const uint4*)&Ds[r * 264 + c];
    }
}

// ---------------------------------------------------------------------------
// Per-edge message + scatter-add: msg[e,o] = sum_i x[src[e],i] * We[e,i,o]
// One wave32 per edge; lane handles outputs {2*lane, 2*lane+1} (packed bf16).
// ---------------------------------------------------------------------------
__global__ void __launch_bounds__(256) conv_msg_kernel(
    const float* __restrict__ x, const u16* __restrict__ Wedge,
    const int* __restrict__ src, const int* __restrict__ dst,
    float* __restrict__ agg)
{
    __shared__ float xs[8][64];
    const int lane = threadIdx.x & 31;
    const int w    = threadIdx.x >> 5;
    const int e    = blockIdx.x * 8 + w;

    const int s = src[e];
    xs[w][lane]      = x[s * 64 + lane];
    xs[w][lane + 32] = x[s * 64 + lane + 32];
    __syncthreads();

    const u16* We = Wedge + (size_t)e * 4096 + 2 * lane;
    float a0 = 0.0f, a1 = 0.0f;
    #pragma unroll 8
    for (int i = 0; i < 64; i++) {
        const u32 u = *(const u32*)(We + i * 64);
        const float f0 = __uint_as_float((u & 0xFFFFu) << 16);
        const float f1 = __uint_as_float(u & 0xFFFF0000u);
        const float xv = xs[w][i];
        a0 = fmaf(xv, f0, a0);
        a1 = fmaf(xv, f1, a1);
    }
    const int d = dst[e];
    atomicAdd(&agg[d * 64 + 2 * lane + 0], a0);
    atomicAdd(&agg[d * 64 + 2 * lane + 1], a1);
}

// ---------------------------------------------------------------------------
// out = agg + x @ W_root + bias  (+ optional GELU)
// ---------------------------------------------------------------------------
__global__ void __launch_bounds__(256) root_kernel(
    const float* __restrict__ xin, const float* __restrict__ agg,
    const float* __restrict__ Wr, const float* __restrict__ bias,
    float* __restrict__ xout, int apply_gelu)
{
    const int t = blockIdx.x * blockDim.x + threadIdx.x;
    const int n = t >> 6;
    const int o = t & 63;
    float v = agg[t] + bias[o];
    float s = 0.0f;
    #pragma unroll 8
    for (int i = 0; i < 64; i++) s = fmaf(xin[n * 64 + i], Wr[i * 64 + o], s);
    v += s;
    if (apply_gelu) v = gelu_f(v);
    xout[t] = v;
}

// ---------------------------------------------------------------------------
extern "C" void kernel_launch(void* const* d_in, const int* in_sizes, int n_in,
                              void* d_out, int out_size, void* d_ws, size_t ws_size,
                              hipStream_t stream)
{
    const float* nodes     = (const float*)d_in[0];
    const int*   edge_idx  = (const int*)  d_in[1];
    const float* edge_attr = (const float*)d_in[2];
    const float* W1 = (const float*)d_in[3];  const float* b1 = (const float*)d_in[4];
    const float* W2 = (const float*)d_in[5];  const float* b2 = (const float*)d_in[6];
    const float* W3 = (const float*)d_in[7];  const float* b3 = (const float*)d_in[8];
    const float* Wr = (const float*)d_in[9];  const float* bias = (const float*)d_in[10];
    const int* src = edge_idx;
    const int* dst = edge_idx + NEDGES;

    char* ws = (char*)d_ws;
    size_t off = 0;
    auto alloc = [&](size_t bytes) -> void* {
        void* p = ws + off;
        off = (off + bytes + 255) & ~(size_t)255;
        return p;
    };
    u16*   W2t   = (u16*)  alloc((size_t)256 * 256 * 2);        // 128 KB
    u16*   W3t   = (u16*)  alloc((size_t)4096 * 256 * 2);       // 2 MB
    u16*   h2    = (u16*)  alloc((size_t)NEDGES * 256 * 2);     // 32 MB
    u16*   Wedge = (u16*)  alloc((size_t)NEDGES * 4096 * 2);    // 512 MB
    float* x1    = (float*)alloc((size_t)NNODES * 64 * 4);      // 1 MB
    float* agg   = (float*)alloc((size_t)NNODES * 64 * 4);      // 1 MB
    (void)ws_size; (void)in_sizes; (void)n_in; (void)out_size;

    convert_weights_kernel<<<512, 256, 0, stream>>>(W2, W3, W2t, W3t);
    mlp12_kernel<<<NEDGES / 64, 256, 0, stream>>>(edge_attr, W1, b1, W2t, b2, h2);

    dim3 g2(NEDGES / 64, 4096 / 256);
    wedge_gemm_kernel<<<g2, 256, 0, stream>>>(h2, W3t, b3, Wedge);

    float* out = (float*)d_out;

    // layer 0
    hipMemsetAsync(agg, 0, (size_t)NNODES * 64 * 4, stream);
    conv_msg_kernel<<<NEDGES / 8, 256, 0, stream>>>(nodes, Wedge, src, dst, agg);
    root_kernel<<<NNODES * 64 / 256, 256, 0, stream>>>(nodes, agg, Wr, bias, x1, 1);

    // layer 1 (no trailing GELU)
    hipMemsetAsync(agg, 0, (size_t)NNODES * 64 * 4, stream);
    conv_msg_kernel<<<NEDGES / 8, 256, 0, stream>>>(x1, Wedge, src, dst, agg);
    root_kernel<<<NNODES * 64 / 256, 256, 0, stream>>>(x1, agg, Wr, bias, out, 0);
}
